// MultiHeadGraphConvLayer_19628000542986
// MI455X (gfx1250) — matrix operand order repaired and checked by reading, not compiled, not run on hardware
//
#include <hip/hip_runtime.h>
#include <hip/hip_bf16.h>
#include <float.h>

typedef __attribute__((ext_vector_type(16))) __bf16 bf16x16;
typedef __attribute__((ext_vector_type(8)))  float  f32x8;

#define WAVES_PER_BLOCK 4
#define TB (WAVES_PER_BLOCK * 32)

// ---------- fragment builders (CDNA5 wave32 WMMA layouts, CDNA5 ISA §7.12.2) ----------

// A-matrix 16x32 bf16: lane L holds row M=L&15; lanes 0-15: K=0..7,16..23 ; lanes 16-31: K=8..15,24..31
__device__ __forceinline__ bf16x16 a_from_global(const float* rowp, int kbase) {
  bf16x16 a;
#pragma unroll
  for (int e = 0; e < 8; ++e) a[e] = (__bf16)rowp[kbase + e];
#pragma unroll
  for (int e = 0; e < 8; ++e) a[8 + e] = (__bf16)rowp[16 + kbase + e];
  return a;
}

__device__ __forceinline__ bf16x16 a_from_lds(const __bf16* h, int M, int k0, int kbase) {
  bf16x16 a;
#pragma unroll
  for (int e = 0; e < 8; ++e) a[e] = h[M * 64 + k0 + kbase + e];
#pragma unroll
  for (int e = 0; e < 8; ++e) a[8 + e] = h[M * 64 + k0 + 16 + kbase + e];
  return a;
}

// B-matrix 32x16 bf16: lane L holds column N=L&15; lanes 0-15: K=0..7,16..23 ; lanes 16-31: K=8..15,24..31
__device__ __forceinline__ bf16x16 b_from_lds(const __bf16* w, int krow, int ldn, int n0) {
  bf16x16 b;
#pragma unroll
  for (int e = 0; e < 16; ++e) b[e] = w[((krow & ~31) + ((krow >> 4) & 1) * 8 + (e < 8 ? e : e + 8)) * ldn + n0 + (krow & 15)];
  return b;
}

__device__ __forceinline__ f32x8 wmma_bf16(bf16x16 a, bf16x16 b, f32x8 c) {
  return __builtin_amdgcn_wmma_f32_16x16x32_bf16(false, a, false, b, (short)0, c, false, false);
}

// ---------- kernel 1: init accumulators ----------
__global__ void k_init(float* segmax, float* denom, float* agg, int N) {
  int i = blockIdx.x * blockDim.x + threadIdx.x;
  if (i < N * 8)  { segmax[i] = -FLT_MAX; denom[i] = 0.0f; }
  if (i < N * 64) { agg[i] = 0.0f; }
}

// ---------- kernel 2: v = atom_feat @ Wv + bv  (N x 8) ----------
__global__ void k_values(const float* __restrict__ atom, const float* __restrict__ Wv,
                         const float* __restrict__ bv, float* __restrict__ v, int N) {
  int i = blockIdx.x * blockDim.x + threadIdx.x;
  if (i >= N * 8) return;
  int n = i >> 3, dv = i & 7;
  float s = bv[dv];
  const float* row = atom + (size_t)n * 64;
#pragma unroll
  for (int k = 0; k < 64; ++k) s += row[k] * Wv[k * 8 + dv];
  v[i] = s;
}

// ---------- kernel 3: edge MLP (3 GEMMs) + logits + segment max ----------
__global__ __launch_bounds__(TB) void k_edge_mlp(
    const float* __restrict__ atom, const float* __restrict__ bond,
    const int* __restrict__ eidx,
    const float* __restrict__ W1, const float* __restrict__ b1,
    const float* __restrict__ W2, const float* __restrict__ b2,
    const float* __restrict__ W3, const float* __restrict__ b3,
    float* __restrict__ logits, float* __restrict__ segmax, int ETILES) {
  __shared__ __bf16 sW1[192 * 64];                  // 24 KB
  __shared__ __bf16 sW2[64 * 64];                   //  8 KB
  __shared__ __bf16 sW3[64 * 16];                   //  2 KB (N padded 8->16 with zeros)
  __shared__ __bf16 sH1[WAVES_PER_BLOCK][16 * 64];  //  8 KB
  __shared__ __bf16 sH2[WAVES_PER_BLOCK][16 * 64];  //  8 KB

  for (int i = threadIdx.x; i < 192 * 64; i += TB) sW1[i] = (__bf16)W1[i];
  for (int i = threadIdx.x; i < 64 * 64; i += TB)  sW2[i] = (__bf16)W2[i];
  for (int i = threadIdx.x; i < 64 * 16; i += TB) {
    int k = i >> 4, n = i & 15;
    sW3[i] = (n < 8) ? (__bf16)W3[k * 8 + n] : (__bf16)0.0f;
  }
  __syncthreads();

  const int wave = threadIdx.x >> 5;
  const int lane = threadIdx.x & 31;
  const int M = lane & 15;            // row within tile (A side) / column N (C side)
  const int half = lane >> 4;
  const int kbase = half * 8;
  const int krow = (lane & 15) + 16 * half;  // B-side K row within 32-K tile

  const int t = blockIdx.x * WAVES_PER_BLOCK + wave;
  if (t >= ETILES) return;  // wave-uniform -> EXEC all ones inside

  const int em = t * 16 + M;
  const int srcI = eidx[2 * em + 0];
  const int dstI = eidx[2 * em + 1];
  __bf16* h1 = sH1[wave];
  __bf16* h2 = sH2[wave];

  // ---- layer 1: relu(x_att @ W1 + b1), x_att = [atom[dst] | atom[src] | bond] (K=192)
#pragma unroll
  for (int nt = 0; nt < 4; ++nt) {
    float bias = b1[nt * 16 + (lane & 15)];
    f32x8 acc = {bias, bias, bias, bias, bias, bias, bias, bias};
#pragma unroll
    for (int kt = 0; kt < 6; ++kt) {
      const float* rowp;
      if (kt < 2)      rowp = atom + (size_t)dstI * 64 + kt * 32;
      else if (kt < 4) rowp = atom + (size_t)srcI * 64 + (kt - 2) * 32;
      else             rowp = bond + (size_t)em * 64 + (kt - 4) * 32;
      bf16x16 a = a_from_global(rowp, kbase);
      bf16x16 b = b_from_lds(sW1, kt * 32 + krow, 64, nt * 16);
      acc = wmma_bf16(a, b, acc);
    }
#pragma unroll
    for (int r = 0; r < 8; ++r) {
      int Mr = r + 8 * half;
      float val = acc[r];
      h1[Mr * 64 + nt * 16 + (lane & 15)] = (__bf16)(val > 0.0f ? val : 0.0f);
    }
  }

  // ---- layer 2: relu(h1 @ W2 + b2) (K=64)
#pragma unroll
  for (int nt = 0; nt < 4; ++nt) {
    float bias = b2[nt * 16 + (lane & 15)];
    f32x8 acc = {bias, bias, bias, bias, bias, bias, bias, bias};
#pragma unroll
    for (int kt = 0; kt < 2; ++kt) {
      bf16x16 a = a_from_lds(h1, M, kt * 32, kbase);
      bf16x16 b = b_from_lds(sW2, kt * 32 + krow, 64, nt * 16);
      acc = wmma_bf16(a, b, acc);
    }
#pragma unroll
    for (int r = 0; r < 8; ++r) {
      int Mr = r + 8 * half;
      float val = acc[r];
      h2[Mr * 64 + nt * 16 + (lane & 15)] = (__bf16)(val > 0.0f ? val : 0.0f);
    }
  }

  // ---- layer 3: logits = h2 @ W3 + b3 (K=64, N=8 padded to 16)
  {
    int n = lane & 15;
    float bias = (n < 8) ? b3[n] : 0.0f;
    f32x8 acc = {bias, bias, bias, bias, bias, bias, bias, bias};
#pragma unroll
    for (int kt = 0; kt < 2; ++kt) {
      bf16x16 a = a_from_lds(h2, M, kt * 32, kbase);
      bf16x16 b = b_from_lds(sW3, kt * 32 + krow, 16, 0);
      acc = wmma_bf16(a, b, acc);
    }
    if (n < 8) {
#pragma unroll
      for (int r = 0; r < 8; ++r) {
        int Mr = r + 8 * half;
        int er = t * 16 + Mr;
        float val = acc[r];
        logits[(size_t)er * 8 + n] = val;
        int s = eidx[2 * er];
        __hip_atomic_fetch_max(&segmax[(size_t)s * 8 + n], val,
                               __ATOMIC_RELAXED, __HIP_MEMORY_SCOPE_AGENT);
      }
    }
  }
}

// ---------- kernel 4: e = exp(logit - segmax[src]); denom += e ----------
__global__ void k_expsum(const int* __restrict__ eidx, float* __restrict__ lg,
                         const float* __restrict__ segmax, float* __restrict__ denom, int E) {
  int i = blockIdx.x * blockDim.x + threadIdx.x;
  if (i >= E * 8) return;
  int e = i >> 3, h = i & 7;
  int s = eidx[2 * e];
  float ev = __expf(lg[i] - segmax[s * 8 + h]);
  lg[i] = ev;
  atomicAdd(&denom[s * 8 + h], ev);
}

// ---------- kernel 5: agg[src] += (e/denom[src]) outer v[dst] ----------
__global__ void k_scatter(const int* __restrict__ eidx, const float* __restrict__ ebuf,
                          const float* __restrict__ denom, const float* __restrict__ v,
                          float* __restrict__ agg, long total) {
  long i = (long)blockIdx.x * blockDim.x + threadIdx.x;
  if (i >= total) return;
  int e = (int)(i >> 6);
  int j = (int)(i & 63);
  int dv = j >> 3, h = j & 7;
  int s = eidx[2 * e], d = eidx[2 * e + 1];
  float att = ebuf[(size_t)e * 8 + h] / denom[(size_t)s * 8 + h];
  float msg = att * v[(size_t)d * 8 + dv];
  atomicAdd(&agg[(size_t)s * 64 + j], msg);
}

// ---------- kernel 6: out_pre = agg @ Wc + bc ; out = relu(atom + out_pre) ----------
__global__ __launch_bounds__(TB) void k_node_out(
    const float* __restrict__ agg, const float* __restrict__ atom,
    const float* __restrict__ Wc, const float* __restrict__ bc,
    float* __restrict__ out_pre, float* __restrict__ out, int NTILES) {
  __shared__ __bf16 sWc[64 * 64];  // 8 KB
  for (int i = threadIdx.x; i < 64 * 64; i += TB) sWc[i] = (__bf16)Wc[i];
  __syncthreads();

  const int wave = threadIdx.x >> 5;
  const int lane = threadIdx.x & 31;
  const int M = lane & 15;
  const int half = lane >> 4;
  const int kbase = half * 8;
  const int krow = (lane & 15) + 16 * half;

  const int t = blockIdx.x * WAVES_PER_BLOCK + wave;
  if (t >= NTILES) return;
  const int node_m = t * 16 + M;
  const float* arow = agg + (size_t)node_m * 64;

#pragma unroll
  for (int nt = 0; nt < 4; ++nt) {
    float bias = bc[nt * 16 + (lane & 15)];
    f32x8 acc = {bias, bias, bias, bias, bias, bias, bias, bias};
#pragma unroll
    for (int kt = 0; kt < 2; ++kt) {
      bf16x16 a = a_from_global(arow + kt * 32, kbase);
      bf16x16 b = b_from_lds(sWc, kt * 32 + krow, 64, nt * 16);
      acc = wmma_bf16(a, b, acc);
    }
#pragma unroll
    for (int r = 0; r < 8; ++r) {
      int node = t * 16 + r + 8 * half;
      int col = nt * 16 + (lane & 15);
      float val = acc[r];
      out_pre[(size_t)node * 64 + col] = val;
      float o = atom[(size_t)node * 64 + col] + val;
      out[(size_t)node * 64 + col] = o > 0.0f ? o : 0.0f;
    }
  }
}

// ---------- kernel 7: new_bond = relu([out_pre[dst]+out_pre[src] | bond] @ Wb + bb) ----------
__global__ __launch_bounds__(TB) void k_bond_out(
    const float* __restrict__ out_pre, const float* __restrict__ bond,
    const int* __restrict__ eidx,
    const float* __restrict__ Wb, const float* __restrict__ bb,
    float* __restrict__ new_bond, int ETILES) {
  __shared__ __bf16 sWb[128 * 64];  // 16 KB
  for (int i = threadIdx.x; i < 128 * 64; i += TB) sWb[i] = (__bf16)Wb[i];
  __syncthreads();

  const int wave = threadIdx.x >> 5;
  const int lane = threadIdx.x & 31;
  const int M = lane & 15;
  const int half = lane >> 4;
  const int kbase = half * 8;
  const int krow = (lane & 15) + 16 * half;

  const int t = blockIdx.x * WAVES_PER_BLOCK + wave;
  if (t >= ETILES) return;
  const int em = t * 16 + M;
  const int srcI = eidx[2 * em + 0];
  const int dstI = eidx[2 * em + 1];
  const float* dr = out_pre + (size_t)dstI * 64;
  const float* sr = out_pre + (size_t)srcI * 64;
  const float* br = bond + (size_t)em * 64;

#pragma unroll
  for (int nt = 0; nt < 4; ++nt) {
    float bias = bb[nt * 16 + (lane & 15)];
    f32x8 acc = {bias, bias, bias, bias, bias, bias, bias, bias};
#pragma unroll
    for (int kt = 0; kt < 4; ++kt) {
      bf16x16 a;
      if (kt < 2) {
        int k0 = kt * 32;
#pragma unroll
        for (int e2 = 0; e2 < 8; ++e2) a[e2] = (__bf16)(dr[k0 + kbase + e2] + sr[k0 + kbase + e2]);
#pragma unroll
        for (int e2 = 0; e2 < 8; ++e2) a[8 + e2] = (__bf16)(dr[k0 + 16 + kbase + e2] + sr[k0 + 16 + kbase + e2]);
      } else {
        a = a_from_global(br + (kt - 2) * 32, kbase);
      }
      bf16x16 b = b_from_lds(sWb, kt * 32 + krow, 64, nt * 16);
      acc = wmma_bf16(a, b, acc);
    }
#pragma unroll
    for (int r = 0; r < 8; ++r) {
      int er = t * 16 + r + 8 * half;
      int col = nt * 16 + (lane & 15);
      float val = acc[r];
      new_bond[(size_t)er * 64 + col] = val > 0.0f ? val : 0.0f;
    }
  }
}

extern "C" void kernel_launch(void* const* d_in, const int* in_sizes, int n_in,
                              void* d_out, int out_size, void* d_ws, size_t ws_size,
                              hipStream_t stream) {
  const float* atom = (const float*)d_in[0];
  const float* bond = (const float*)d_in[1];
  const int*   eidx = (const int*)d_in[2];
  const float* Wv = (const float*)d_in[3];
  const float* bv = (const float*)d_in[4];
  const float* W1 = (const float*)d_in[5];
  const float* b1 = (const float*)d_in[6];
  const float* W2 = (const float*)d_in[7];
  const float* b2 = (const float*)d_in[8];
  const float* W3 = (const float*)d_in[9];
  const float* b3 = (const float*)d_in[10];
  const float* Wc = (const float*)d_in[11];
  const float* bc = (const float*)d_in[12];
  const float* Wb = (const float*)d_in[13];
  const float* bb = (const float*)d_in[14];

  const int N = in_sizes[0] / 64;
  const int E = in_sizes[1] / 64;
  const int ETILES = (E + 15) / 16;
  const int NTILES = (N + 15) / 16;

  float* ws = (float*)d_ws;
  size_t o = 0;
  float* v       = ws + o; o += (size_t)N * 8;
  float* lg      = ws + o; o += (size_t)E * 8;   // logits, then exp(e) in-place
  float* segmax  = ws + o; o += (size_t)N * 8;
  float* denom   = ws + o; o += (size_t)N * 8;
  float* agg     = ws + o; o += (size_t)N * 64;
  float* out_pre = ws + o; o += (size_t)N * 64;

  float* out_nodes = (float*)d_out;             // (N, 64)
  float* out_bonds = (float*)d_out + (size_t)N * 64;  // (E, 64)

  // 1. init accumulators
  {
    int total = N * 64;
    k_init<<<(total + 255) / 256, 256, 0, stream>>>(segmax, denom, agg, N);
  }
  // 2. values v = atom @ Wv + bv
  {
    int total = N * 8;
    k_values<<<(total + 255) / 256, 256, 0, stream>>>(atom, Wv, bv, v, N);
  }
  // 3. edge MLP + segment max (WMMA)
  k_edge_mlp<<<(ETILES + WAVES_PER_BLOCK - 1) / WAVES_PER_BLOCK, TB, 0, stream>>>(
      atom, bond, eidx, W1, b1, W2, b2, W3, b3, lg, segmax, ETILES);
  // 4. exp + denom
  {
    int total = E * 8;
    k_expsum<<<(total + 255) / 256, 256, 0, stream>>>(eidx, lg, segmax, denom, E);
  }
  // 5. attention-weighted outer-product scatter
  {
    long total = (long)E * 64;
    k_scatter<<<(unsigned)((total + 255) / 256), 256, 0, stream>>>(eidx, lg, denom, v, agg, total);
  }
  // 6. node output GEMM (WMMA)
  k_node_out<<<(NTILES + WAVES_PER_BLOCK - 1) / WAVES_PER_BLOCK, TB, 0, stream>>>(
      agg, atom, Wc, bc, out_pre, out_nodes, NTILES);
  // 7. bond output GEMM (WMMA)
  k_bond_out<<<(ETILES + WAVES_PER_BLOCK - 1) / WAVES_PER_BLOCK, TB, 0, stream>>>(
      out_pre, bond, eidx, Wb, bb, out_bonds, ETILES);
}
